// GPTModel_66168266162308
// MI455X (gfx1250) — matrix operand order repaired and checked by reading, not compiled, not run on hardware
//
#include <hip/hip_runtime.h>
#include <math.h>
#include <stdint.h>

// ---- problem dims (fixed by reference) ----
#define L_  4
#define B_  4
#define T_  1024
#define C_  1024
#define H_  16
#define DH_ 64
#define FF_ 4096
#define V_  32000
#define BT_ (B_*T_)
#define EPS_ 1e-5f
#define IGNORE_ 5

// CDNA5 async global->LDS staging (ASYNCcnt path). Set to 0 if toolchain rejects.
#define USE_ASYNC_LDS 1

#if USE_ASYNC_LDS
#define ASYNC_WAIT() asm volatile("s_wait_asynccnt 0x0" ::: "memory")
#else
#define ASYNC_WAIT()
#endif

typedef __bf16 bf16;
typedef __attribute__((ext_vector_type(16))) __bf16 v16bf;
typedef __attribute__((ext_vector_type(8)))  __bf16 v8bf;
typedef __attribute__((ext_vector_type(8)))  float  v8f;

union V16u { v16bf v; v8bf h[2]; };
union V8u  { v8bf  v; bf16 e[8]; };

// ============================================================
// f32 W[K][N] -> bf16 Wt[N][K]  (tiled transpose through LDS)
// grid: (N/64, K/64, nMat), block 256
// ============================================================
__global__ __launch_bounds__(256)
void cvt_transpose_kernel(const float* __restrict__ src, bf16* __restrict__ dst,
                          int K, int N) {
    __shared__ bf16 tile[64][72];
    const int l = blockIdx.z;
    const int n0 = blockIdx.x * 64, k0 = blockIdx.y * 64;
    const int r = threadIdx.x >> 2, c0 = (threadIdx.x & 3) * 16;

    const float* sr = src + ((size_t)l * K + k0 + r) * N + n0 + c0;
#pragma unroll
    for (int j = 0; j < 16; ++j) tile[r][c0 + j] = (bf16)sr[j];
    __syncthreads();

    bf16* dr = dst + ((size_t)l * N + n0 + r) * K + k0 + c0;
    V8u o0, o1;
#pragma unroll
    for (int j = 0; j < 8; ++j) { o0.e[j] = tile[c0 + j][r]; o1.e[j] = tile[c0 + 8 + j][r]; }
    *(v8bf*)dr = o0.v;
    *(v8bf*)(dr + 8) = o1.v;
}

// ============================================================
// embedding: x[row][c] = tok_emb[idx[row]][c] + pos_emb[row%T][c]
// ============================================================
__global__ __launch_bounds__(256) void embed_kernel(const int* __restrict__ idx,
                                                    const float* __restrict__ tok,
                                                    const float* __restrict__ pos,
                                                    float* __restrict__ x) {
    int row = blockIdx.x, tid = threadIdx.x;
    int t = row & (T_ - 1);
    int id = idx[row];
    const float* ts = tok + (size_t)id * C_;
    const float* ps = pos + (size_t)t * C_;
    float* xr = x + (size_t)row * C_;
#pragma unroll
    for (int i = 0; i < 4; ++i) {
        int c = tid + i * 256;
        xr[c] = ts[c] + ps[c];
    }
}

// ============================================================
// layernorm (one row per block), bf16 output
// ============================================================
__global__ __launch_bounds__(256) void ln_bf16_kernel(const float* __restrict__ x,
                                                      const float* __restrict__ g,
                                                      const float* __restrict__ b,
                                                      bf16* __restrict__ out) {
    __shared__ float s1[256], s2[256];
    int row = blockIdx.x, tid = threadIdx.x;
    const float* xr = x + (size_t)row * C_;
    float v[4], sum = 0.f, sq = 0.f;
#pragma unroll
    for (int i = 0; i < 4; ++i) { v[i] = xr[tid + i * 256]; sum += v[i]; sq += v[i] * v[i]; }
    s1[tid] = sum; s2[tid] = sq; __syncthreads();
    for (int st = 128; st > 0; st >>= 1) {
        if (tid < st) { s1[tid] += s1[tid + st]; s2[tid] += s2[tid + st]; }
        __syncthreads();
    }
    float mu = s1[0] * (1.f / C_);
    float var = s2[0] * (1.f / C_) - mu * mu;
    float r = rsqrtf(var + EPS_);
    bf16* orow = out + (size_t)row * C_;
#pragma unroll
    for (int i = 0; i < 4; ++i) {
        int c = tid + i * 256;
        orow[c] = (bf16)((v[i] - mu) * r * g[c] + b[c]);
    }
}

// ============================================================
// bf16 WMMA GEMM: D = A(MxK) @ Bt(NxK, pre-transposed) [+bias][+epilogue]
// 256 thr = 8 waves, tile 128x128x32, wave grid 2(M)x4(N),
// double-buffered LDS, async global->LDS staging (ASYNCcnt).
// mode 0: bf16 out            mode 1: bf16 out, gelu(acc+bias)
// mode 2: f32 Cf += acc+bias  mode 3: f32 Cf = acc+bias
// ============================================================
#define LDT 40   // LDS row stride (32 + 8 pad) in bf16; 80B = 16B multiple

__global__ __launch_bounds__(256)
void gemm_bf16_wmma(const bf16* __restrict__ A, const bf16* __restrict__ Bt,
                    float* __restrict__ Cf, bf16* __restrict__ Cb,
                    const float* __restrict__ bias,
                    int M, int N, int K, int mode) {
    __shared__ bf16 As[2][128 * LDT];
    __shared__ bf16 Bs[2][128 * LDT];
    const int tid = threadIdx.x, lane = tid & 31, wid = tid >> 5;
    const int waveM = wid >> 2, waveN = wid & 3;
    const int mBase = blockIdx.y * 128, nBase = blockIdx.x * 128;
    const int m = lane & 15, half = lane >> 4;
    (void)M;

    // staging map: thread -> (row 0..127, 16-elem k chunk)
    const int srow = tid >> 1, skc = (tid & 1) * 16;
    const bf16* gA = A  + (size_t)(mBase + srow) * K + skc;
    const bf16* gB = Bt + (size_t)(nBase + srow) * K + skc;   // Bt rows are N, cols K

    v8f acc[4][2];
    const v8f vzero = {0.f, 0.f, 0.f, 0.f, 0.f, 0.f, 0.f, 0.f};
#pragma unroll
    for (int i = 0; i < 4; ++i)
#pragma unroll
        for (int j = 0; j < 2; ++j) acc[i][j] = vzero;

    auto stage = [&](int buf, int k0) {
#if USE_ASYNC_LDS
        uint32_t la = (uint32_t)(uintptr_t)&As[buf][srow * LDT + skc];
        uint32_t lb = (uint32_t)(uintptr_t)&Bs[buf][srow * LDT + skc];
        const bf16* pa = gA + k0;
        const bf16* pb = gB + k0;
        asm volatile("global_load_async_to_lds_b128 %0, %1, off" :: "v"(la),       "v"(pa)     : "memory");
        asm volatile("global_load_async_to_lds_b128 %0, %1, off" :: "v"(la + 16u), "v"(pa + 8) : "memory");
        asm volatile("global_load_async_to_lds_b128 %0, %1, off" :: "v"(lb),       "v"(pb)     : "memory");
        asm volatile("global_load_async_to_lds_b128 %0, %1, off" :: "v"(lb + 16u), "v"(pb + 8) : "memory");
#else
        *(v8bf*)&As[buf][srow * LDT + skc]     = *(const v8bf*)(gA + k0);
        *(v8bf*)&As[buf][srow * LDT + skc + 8] = *(const v8bf*)(gA + k0 + 8);
        *(v8bf*)&Bs[buf][srow * LDT + skc]     = *(const v8bf*)(gB + k0);
        *(v8bf*)&Bs[buf][srow * LDT + skc + 8] = *(const v8bf*)(gB + k0 + 8);
#endif
    };

    const int nk = K >> 5;
    stage(0, 0);
    ASYNC_WAIT();
    __syncthreads();

    for (int kt = 0; kt < nk; ++kt) {
        const int cur = kt & 1;
        if (kt + 1 < nk) stage(cur ^ 1, (kt + 1) << 5);

        V16u af[4], bfg[2];
#pragma unroll
        for (int i = 0; i < 4; ++i) {
            const bf16* p = &As[cur][(waveM * 64 + i * 16 + m) * LDT];
            af[i].h[0] = *(const v8bf*)(p + half * 8);        // K 0-7  / 8-15
            af[i].h[1] = *(const v8bf*)(p + 16 + half * 8);   // K 16-23/ 24-31
        }
#pragma unroll
        for (int j = 0; j < 2; ++j) {
            const bf16* p = &Bs[cur][(waveN * 32 + j * 16 + m) * LDT];
            bfg[j].h[0] = *(const v8bf*)(p + half * 8);       // K 0-7  / 8-15
            bfg[j].h[1] = *(const v8bf*)(p + 16 + half * 8);  // K 16-23/ 24-31
        }
#pragma unroll
        for (int i = 0; i < 4; ++i)
#pragma unroll
            for (int j = 0; j < 2; ++j)
                acc[i][j] = __builtin_amdgcn_wmma_f32_16x16x32_bf16(
                    false, af[i].v, false, bfg[j].v, (short)0, acc[i][j], false, false);

        ASYNC_WAIT();       // next tile resident in other buffer
        __syncthreads();    // all waves' staging visible
    }

    // ---- epilogue (C layout: VGPR r -> row r + 8*half, col = lane&15) ----
#pragma unroll
    for (int i = 0; i < 4; ++i) {
#pragma unroll
        for (int j = 0; j < 2; ++j) {
            int gm0 = mBase + waveM * 64 + i * 16 + 8 * half;
            int gn = nBase + waveN * 32 + j * 16 + m;
            float bs = (mode != 0) ? bias[gn] : 0.f;
#pragma unroll
            for (int r = 0; r < 8; ++r) {
                float val = acc[i][j][r];
                size_t off = (size_t)(gm0 + r) * N + gn;
                if (mode == 0) {
                    Cb[off] = (bf16)val;
                } else if (mode == 1) {
                    float u = val + bs;
                    Cb[off] = (bf16)(0.5f * u * (1.f + erff(u * 0.70710678118654752f)));
                } else if (mode == 2) {
                    Cf[off] += val + bs;
                } else {
                    Cf[off] = val + bs;
                }
            }
        }
    }
}

// ============================================================
// flash attention: one block per (b, head, 128-row q tile)
// 8 waves, each owns 16 q rows end-to-end (S, softmax state, O)
// ============================================================
#define LDQ 72   // 64 + 8 pad; 144B row stride

__global__ __launch_bounds__(256)
void attn_kernel(const bf16* __restrict__ q, const bf16* __restrict__ k,
                 const bf16* __restrict__ v, bf16* __restrict__ att) {
    __shared__ bf16 Qs[128 * LDQ];
    __shared__ bf16 Ks[64 * LDQ];    // row = k-time, col = d (B operand as-is)
    __shared__ bf16 Vs[64 * LDQ];    // transposed: row = d, col = k-time
    __shared__ bf16 Ps[128 * LDQ];
    const int tid = threadIdx.x, lane = tid & 31, wid = tid >> 5;
    const int qBase = blockIdx.x * 128;
    const int head = blockIdx.y, bb = blockIdx.z;
    const int m = lane & 15, half = lane >> 4;
    const size_t base = ((size_t)bb * T_) * C_ + head * DH_;

    {   // load Q tile 128x64
        int r = tid >> 1, d0 = (tid & 1) * 32;
        const bf16* src = q + base + (size_t)(qBase + r) * C_ + d0;
#pragma unroll
        for (int j = 0; j < 4; ++j)
            *(v8bf*)&Qs[r * LDQ + d0 + j * 8] = *(const v8bf*)(src + j * 8);
    }

    float rm[8], rs[8];
#pragma unroll
    for (int i = 0; i < 8; ++i) { rm[i] = -1e30f; rs[i] = 0.f; }
    const v8f vzero = {0.f, 0.f, 0.f, 0.f, 0.f, 0.f, 0.f, 0.f};
    v8f oacc[4];
#pragma unroll
    for (int j = 0; j < 4; ++j) oacc[j] = vzero;

    __syncthreads();

    const int ktMax = (qBase >> 6) + 2;   // causal
    for (int kt = 0; kt < ktMax; ++kt) {
        const int kBase = kt * 64;
        {   // stage K (as-is) and V (transposed)
            int r = tid >> 2, d0 = (tid & 3) * 16;
            const bf16* ksrc = k + base + (size_t)(kBase + r) * C_ + d0;
            *(v8bf*)&Ks[r * LDQ + d0]     = *(const v8bf*)(ksrc);
            *(v8bf*)&Ks[r * LDQ + d0 + 8] = *(const v8bf*)(ksrc + 8);
            const bf16* vsrc = v + base + (size_t)(kBase + r) * C_ + d0;
            V8u a0, a1;
            a0.v = *(const v8bf*)vsrc;
            a1.v = *(const v8bf*)(vsrc + 8);
#pragma unroll
            for (int j = 0; j < 8; ++j) {
                Vs[(d0 + j) * LDQ + r]     = a0.e[j];
                Vs[(d0 + 8 + j) * LDQ + r] = a1.e[j];
            }
        }
        __syncthreads();

        // S = Q K^T for this wave's 16 rows x 64 cols
        v8f sacc[4];
#pragma unroll
        for (int j = 0; j < 4; ++j) sacc[j] = vzero;
#pragma unroll
        for (int ks = 0; ks < 64; ks += 32) {
            V16u a;
            a.h[0] = *(const v8bf*)&Qs[(wid * 16 + m) * LDQ + ks + half * 8];
            a.h[1] = *(const v8bf*)&Qs[(wid * 16 + m) * LDQ + ks + 16 + half * 8];
#pragma unroll
            for (int j = 0; j < 4; ++j) {
                V16u bfr;
                bfr.h[0] = *(const v8bf*)&Ks[(j * 16 + m) * LDQ + ks + half * 8];
                bfr.h[1] = *(const v8bf*)&Ks[(j * 16 + m) * LDQ + ks + 16 + half * 8];
                sacc[j] = __builtin_amdgcn_wmma_f32_16x16x32_bf16(
                    false, a.v, false, bfr.v, (short)0, sacc[j], false, false);
            }
        }

        // online softmax (rows stay inside this wave)
#pragma unroll
        for (int r = 0; r < 8; ++r) {
            const int qq = qBase + wid * 16 + 8 * half + r;
            float mx = rm[r];
            float sv[4];
#pragma unroll
            for (int j = 0; j < 4; ++j) {
                int kk = kBase + j * 16 + m;
                float s = sacc[j][r] * 0.125f;          // DH^-0.5
                s = (kk <= qq) ? s : -1e30f;            // causal mask
                sv[j] = s;
                mx = fmaxf(mx, s);
            }
#pragma unroll
            for (int off = 1; off < 16; off <<= 1) mx = fmaxf(mx, __shfl_xor(mx, off, 32));
            float scale = __expf(rm[r] - mx);
            float lsum = 0.f;
#pragma unroll
            for (int j = 0; j < 4; ++j) {
                float p = __expf(sv[j] - mx);
                lsum += p;
                Ps[(wid * 16 + 8 * half + r) * LDQ + j * 16 + m] = (bf16)p;
            }
#pragma unroll
            for (int off = 1; off < 16; off <<= 1) lsum += __shfl_xor(lsum, off, 32);
            rs[r] = rs[r] * scale + lsum;
            rm[r] = mx;
#pragma unroll
            for (int j = 0; j < 4; ++j) oacc[j][r] *= scale;
        }

        // O += P @ V (same-wave LDS store->load is in-order)
#pragma unroll
        for (int ks = 0; ks < 64; ks += 32) {
            V16u a;
            a.h[0] = *(const v8bf*)&Ps[(wid * 16 + m) * LDQ + ks + half * 8];
            a.h[1] = *(const v8bf*)&Ps[(wid * 16 + m) * LDQ + ks + 16 + half * 8];
#pragma unroll
            for (int j = 0; j < 4; ++j) {
                V16u bfr;
                bfr.h[0] = *(const v8bf*)&Vs[(j * 16 + m) * LDQ + ks + half * 8];
                bfr.h[1] = *(const v8bf*)&Vs[(j * 16 + m) * LDQ + ks + 16 + half * 8];
                oacc[j] = __builtin_amdgcn_wmma_f32_16x16x32_bf16(
                    false, a.v, false, bfr.v, (short)0, oacc[j], false, false);
            }
        }
        __syncthreads();
    }

#pragma unroll
    for (int r = 0; r < 8; ++r) {
        const int qq = qBase + wid * 16 + 8 * half + r;
        float inv = 1.f / rs[r];
#pragma unroll
        for (int j = 0; j < 4; ++j)
            att[base + (size_t)qq * C_ + j * 16 + m] = (bf16)(oacc[j][r] * inv);
    }
}

// ============================================================
// cross-entropy (ignore_index=5): online logsumexp per row
// ============================================================
__global__ void zero_accum_kernel(float* a) { a[0] = 0.f; a[1] = 0.f; }

__global__ __launch_bounds__(256)
void ce_row_kernel(const float* __restrict__ logits, const int* __restrict__ tgt,
                   float* __restrict__ accum) {
    __shared__ float sm[256], ss[256];
    int row = blockIdx.x, tid = threadIdx.x;
    const float* lg = logits + (size_t)row * V_;
    float mval = -INFINITY, s = 0.f;
    for (int i = tid; i < V_; i += 256) {
        float vv = lg[i];
        float nm = fmaxf(mval, vv);
        s = s * __expf(mval - nm) + __expf(vv - nm);
        mval = nm;
    }
    sm[tid] = mval; ss[tid] = s; __syncthreads();
    for (int st = 128; st > 0; st >>= 1) {
        if (tid < st) {
            float m2 = sm[tid + st], s2 = ss[tid + st];
            float nm = fmaxf(sm[tid], m2);
            ss[tid] = ss[tid] * __expf(sm[tid] - nm) + s2 * __expf(m2 - nm);
            sm[tid] = nm;
        }
        __syncthreads();
    }
    if (tid == 0) {
        int t = tgt[row];
        if (t != IGNORE_) {
            float nll = sm[0] + __logf(ss[0]) - lg[t];
            atomicAdd(accum, nll);
            atomicAdd(accum + 1, 1.f);
        }
    }
}

__global__ void finalize_kernel(const float* accum, float* out) {
    out[0] = accum[0] / fmaxf(accum[1], 1.f);
}

// ============================================================
// host-side orchestration
// ============================================================
extern "C" void kernel_launch(void* const* d_in, const int* in_sizes, int n_in,
                              void* d_out, int out_size, void* d_ws, size_t ws_size,
                              hipStream_t stream) {
    const int*   idx     = (const int*)d_in[0];
    const int*   targets = (const int*)d_in[1];
    const float* tok_emb = (const float*)d_in[2];
    const float* pos_emb = (const float*)d_in[3];
    const float* Wq      = (const float*)d_in[4];
    const float* Wk      = (const float*)d_in[5];
    const float* Wv      = (const float*)d_in[6];
    const float* Wo      = (const float*)d_in[7];
    const float* bo      = (const float*)d_in[8];
    const float* ln1_g   = (const float*)d_in[9];
    const float* ln1_b   = (const float*)d_in[10];
    const float* ln2_g   = (const float*)d_in[11];
    const float* ln2_b   = (const float*)d_in[12];
    const float* W1      = (const float*)d_in[13];
    const float* b1      = (const float*)d_in[14];
    const float* W2      = (const float*)d_in[15];
    const float* b2      = (const float*)d_in[16];
    const float* lnf_g   = (const float*)d_in[17];
    const float* lnf_b   = (const float*)d_in[18];
    const float* Whead   = (const float*)d_in[19];
    const float* bhead   = (const float*)d_in[20];

    char* ws = (char*)d_ws;
    size_t off = 0;
    auto alloc = [&](size_t bytes) -> char* {
        char* p = ws + off;
        off = (off + bytes + 255) & ~(size_t)255;
        return p;
    };
    const size_t nCC = (size_t)L_ * C_ * C_;
    const size_t nCF = (size_t)L_ * C_ * FF_;
    bf16* bWqT = (bf16*)alloc(nCC * 2);          // [L][N=C][K=C]
    bf16* bWkT = (bf16*)alloc(nCC * 2);
    bf16* bWvT = (bf16*)alloc(nCC * 2);
    bf16* bWoT = (bf16*)alloc(nCC * 2);
    bf16* bW1T = (bf16*)alloc(nCF * 2);          // [L][N=FF][K=C]
    bf16* bW2T = (bf16*)alloc(nCF * 2);          // [L][N=C][K=FF]
    bf16* bWhT = (bf16*)alloc((size_t)C_ * V_ * 2); // [N=V][K=C]
    float* x  = (float*)alloc((size_t)BT_ * C_ * 4);
    bf16* hb  = (bf16*)alloc((size_t)BT_ * C_ * 2);
    bf16* qb  = (bf16*)alloc((size_t)BT_ * C_ * 2);
    bf16* kb  = (bf16*)alloc((size_t)BT_ * C_ * 2);
    bf16* vb  = (bf16*)alloc((size_t)BT_ * C_ * 2);
    bf16* ab  = (bf16*)alloc((size_t)BT_ * C_ * 2);
    bf16* fb  = (bf16*)alloc((size_t)BT_ * FF_ * 2);
    float* accum = (float*)alloc(256);
    (void)in_sizes; (void)n_in; (void)out_size; (void)ws_size;

    // bf16 + transpose weight cache (rebuilt every call; deterministic)
    cvt_transpose_kernel<<<dim3(C_ / 64, C_ / 64, L_), 256, 0, stream>>>(Wq, bWqT, C_, C_);
    cvt_transpose_kernel<<<dim3(C_ / 64, C_ / 64, L_), 256, 0, stream>>>(Wk, bWkT, C_, C_);
    cvt_transpose_kernel<<<dim3(C_ / 64, C_ / 64, L_), 256, 0, stream>>>(Wv, bWvT, C_, C_);
    cvt_transpose_kernel<<<dim3(C_ / 64, C_ / 64, L_), 256, 0, stream>>>(Wo, bWoT, C_, C_);
    cvt_transpose_kernel<<<dim3(FF_ / 64, C_ / 64, L_), 256, 0, stream>>>(W1, bW1T, C_, FF_);
    cvt_transpose_kernel<<<dim3(C_ / 64, FF_ / 64, L_), 256, 0, stream>>>(W2, bW2T, FF_, C_);
    cvt_transpose_kernel<<<dim3(V_ / 64, C_ / 64, 1), 256, 0, stream>>>(Whead, bWhT, C_, V_);

    embed_kernel<<<BT_, 256, 0, stream>>>(idx, tok_emb, pos_emb, x);

    const dim3 gC(C_ / 128, BT_ / 128);
    const dim3 gF(FF_ / 128, BT_ / 128);
    const dim3 gA(T_ / 128, H_, B_);

    for (int l = 0; l < L_; ++l) {
        ln_bf16_kernel<<<BT_, 256, 0, stream>>>(x, ln1_g + l * C_, ln1_b + l * C_, hb);
        gemm_bf16_wmma<<<gC, 256, 0, stream>>>(hb, bWqT + (size_t)l * C_ * C_, nullptr, qb, nullptr, BT_, C_, C_, 0);
        gemm_bf16_wmma<<<gC, 256, 0, stream>>>(hb, bWkT + (size_t)l * C_ * C_, nullptr, kb, nullptr, BT_, C_, C_, 0);
        gemm_bf16_wmma<<<gC, 256, 0, stream>>>(hb, bWvT + (size_t)l * C_ * C_, nullptr, vb, nullptr, BT_, C_, C_, 0);
        attn_kernel<<<gA, 256, 0, stream>>>(qb, kb, vb, ab);
        gemm_bf16_wmma<<<gC, 256, 0, stream>>>(ab, bWoT + (size_t)l * C_ * C_, x, nullptr, bo + l * C_, BT_, C_, C_, 2);
        ln_bf16_kernel<<<BT_, 256, 0, stream>>>(x, ln2_g + l * C_, ln2_b + l * C_, hb);
        gemm_bf16_wmma<<<gF, 256, 0, stream>>>(hb, bW1T + (size_t)l * C_ * FF_, nullptr, fb, b1 + l * FF_, BT_, FF_, C_, 1);
        gemm_bf16_wmma<<<gC, 256, 0, stream>>>(fb, bW2T + (size_t)l * FF_ * C_, x, nullptr, b2 + l * C_, BT_, C_, FF_, 2);
    }

    ln_bf16_kernel<<<BT_, 256, 0, stream>>>(x, lnf_g, lnf_b, hb);
    float* logits = (float*)d_out;
    gemm_bf16_wmma<<<dim3(V_ / 128, BT_ / 128), 256, 0, stream>>>(hb, bWhT, logits, nullptr, bhead, BT_, V_, C_, 3);

    zero_accum_kernel<<<1, 1, 0, stream>>>(accum);
    ce_row_kernel<<<BT_, 256, 0, stream>>>(logits, targets, accum);
    finalize_kernel<<<1, 1, 0, stream>>>(accum, logits + (size_t)BT_ * V_);
}
